// M9System1_57543971832725
// MI455X (gfx1250) — matrix operand order repaired and checked by reading, not compiled, not run on hardware
//
#include <hip/hip_runtime.h>

typedef __attribute__((ext_vector_type(16))) __bf16 v16bf;
typedef __attribute__((ext_vector_type(8)))  __bf16 v8bf;
typedef __attribute__((ext_vector_type(8)))  float  v8f;

#define HH   896      // hidden dim
#define KC   2000     // codebook entries (125 tiles of 16, exact)
#define NPTR 256      // 2 * MAX_LEN pointer-head rows
#define BB   32768    // batch
#define MT   64       // batch rows per workgroup
#define STR  904      // padded LDS stride (bf16 elements) for bank spread

__device__ __forceinline__ unsigned short f2bf(float f) {
    unsigned u = __float_as_uint(f);
    u += 0x7FFFu + ((u >> 16) & 1u);          // round-to-nearest-even
    return (unsigned short)(u >> 16);
}

// ---- elementwise f32 -> bf16 conversion (vectorized float4) ----
__global__ void convert_bf16(const float* __restrict__ src,
                             unsigned short* __restrict__ dst, int n4) {
    int i = blockIdx.x * blockDim.x + threadIdx.x;
    if (i < n4) {
        float4 v = ((const float4*)src)[i];
        ushort4 o;
        o.x = f2bf(v.x); o.y = f2bf(v.y); o.z = f2bf(v.z); o.w = f2bf(v.w);
        ((ushort4*)dst)[i] = o;
    }
}

// ---- hn[k] = 0.5 * ||emb_k||^2 ----
__global__ void half_norms(const float* __restrict__ emb, float* __restrict__ hn) {
    const int k = blockIdx.x;
    const int tid = threadIdx.x;
    float s = 0.f;
    for (int h = tid; h < HH; h += 256) {
        float v = emb[(size_t)k * HH + h];
        s += v * v;
    }
#pragma unroll
    for (int off = 16; off >= 1; off >>= 1) s += __shfl_xor(s, off, 32);
    __shared__ float red[8];
    if ((tid & 31) == 0) red[tid >> 5] = s;
    __syncthreads();
    if (tid == 0) {
        float t = 0.f;
#pragma unroll
        for (int w = 0; w < 8; ++w) t += red[w];
        hn[k] = 0.5f * t;
    }
}

// 4 M-subtiles x one N-tile GEMM over the full K=896 dim.
// Rolled k-loop (A-loads stay in-loop: no LICM->spill); B-fragment is
// software-pipelined one k-step ahead (32B over-read past the last row is
// safe: emb_bf | w_bf | hn are adjacent in the workspace).
__device__ __forceinline__ void gemm_k(const unsigned short* __restrict__ bp,
                                       const unsigned short* __restrict__ abase,
                                       v8f (&acc)[4]) {
    v16bf bf = *(const v16bf*)bp;
#pragma unroll 1
    for (int k0 = 0; k0 < HH; k0 += 32) {
        v16bf bfn = *(const v16bf*)(bp + k0 + 32);   // prefetch next k-step
#pragma unroll
        for (int m = 0; m < 4; ++m) {
            const unsigned short* ap = abase + m * 16 * STR + k0;
            v8bf a0 = *(const v8bf*)ap;
            v8bf a1 = *(const v8bf*)(ap + 8);
            v16bf af = __builtin_shufflevector(a0, a1,
                0,1,2,3,4,5,6,7,8,9,10,11,12,13,14,15);
            acc[m] = __builtin_amdgcn_wmma_f32_16x16x32_bf16(
                false, af, false, bf, (short)0, acc[m], false, false);
        }
        bf = bfn;
    }
}

// ---- fused: VQ argmin + gather + pointer logits, WMMA bf16 ----
__global__ __launch_bounds__(256, 1) void vq_ptr_kernel(
    const float* __restrict__ z, const float* __restrict__ emb,
    const unsigned short* __restrict__ emb_bf, const float* __restrict__ hn,
    const unsigned short* __restrict__ w_bf, const float* __restrict__ bias,
    const int* __restrict__ plen_p,
    float* __restrict__ outA, float* __restrict__ outB) {

    __shared__ unsigned short zt[MT * STR];    // 115,712 B bf16 z tile
    __shared__ float wbv[8][MT];
    __shared__ int   wbi[8][MT];
    __shared__ int   fidx[MT];

    const int tid = threadIdx.x;
    const int r0  = blockIdx.x * MT;

    // ---- stage 64-row z tile into LDS as bf16 ----
    {
        const float4* src = (const float4*)(z + (size_t)r0 * HH);
        for (int i = tid; i < MT * HH / 4; i += 256) {
            float4 v = src[i];
            int row = i / (HH / 4);
            int c4  = i - row * (HH / 4);
            ushort4 o;
            o.x = f2bf(v.x); o.y = f2bf(v.y); o.z = f2bf(v.z); o.w = f2bf(v.w);
            *(ushort4*)&zt[row * STR + c4 * 4] = o;
        }
    }
    __syncthreads();

    const int wave = tid >> 5;
    const int lane = tid & 31;
    const int half = lane >> 4;     // A/B K-half select (wave32 layout)
    const int ln   = lane & 15;     // A row / B column within tile
    const unsigned short* abase = &zt[ln * STR + half * 16];

    float bv[4][8];
    int   bi[4][8];
#pragma unroll
    for (int m = 0; m < 4; ++m)
#pragma unroll
        for (int i = 0; i < 8; ++i) { bv[m][i] = -__builtin_inff(); bi[m][i] = 0x7fffffff; }

    // ---- distance GEMM: argmax of z.e - 0.5||e||^2 over 125 code tiles ----
#pragma unroll 1
    for (int t = wave; t < KC / 16; t += 8) {
        const int code = t * 16 + ln;
        v8f acc[4];
#pragma unroll
        for (int m = 0; m < 4; ++m)
#pragma unroll
            for (int i = 0; i < 8; ++i) acc[m][i] = 0.f;

        gemm_k(emb_bf + (size_t)code * HH + half * 16, abase, acc);

        // Branchless per-lane argmax update (v_cmp + v_cndmask, no exec churn).
        // Within a lane `code` is strictly increasing across tiles, so a strict
        // ">" implements "first index wins ties" with no explicit tie-break.
        const float hnv = hn[code];
#pragma unroll
        for (int m = 0; m < 4; ++m)
#pragma unroll
            for (int i = 0; i < 8; ++i) {
                float v = acc[m][i] - hnv;
                bool better = v > bv[m][i];
                bv[m][i] = better ? v    : bv[m][i];
                bi[m][i] = better ? code : bi[m][i];
            }
    }

    // ---- argmax reduce across the 16 lanes of each half (wave32 shuffles) ----
    // Tie-break (lower index wins) needed here: lanes hold unordered code sets.
#pragma unroll
    for (int off = 8; off >= 1; off >>= 1) {
#pragma unroll
        for (int m = 0; m < 4; ++m)
#pragma unroll
            for (int i = 0; i < 8; ++i) {
                float ov = __shfl_xor(bv[m][i], off, 32);
                int   oi = __shfl_xor(bi[m][i], off, 32);
                bool better = (ov > bv[m][i]) ||
                              (ov == bv[m][i] && oi < bi[m][i]);
                bv[m][i] = better ? ov : bv[m][i];
                bi[m][i] = better ? oi : bi[m][i];
            }
    }
    if (ln == 0) {
#pragma unroll
        for (int m = 0; m < 4; ++m)
#pragma unroll
            for (int i = 0; i < 8; ++i) {
                int row = m * 16 + half * 8 + i;   // C-layout row mapping
                wbv[wave][row] = bv[m][i];
                wbi[wave][row] = bi[m][i];
            }
    }
    __syncthreads();

    if (tid < MT) {
        float best = wbv[0][tid]; int idx = wbi[0][tid];
#pragma unroll
        for (int w = 1; w < 8; ++w) {
            float v = wbv[w][tid]; int j = wbi[w][tid];
            bool better = (v > best) || (v == best && j < idx);
            best = better ? v : best;
            idx  = better ? j : idx;
        }
        fidx[tid] = idx;
    }
    __syncthreads();

    // ---- op_state = emb[idx] (straight-through forward value) ----
    {
        float4* dst = (float4*)(outA + (size_t)r0 * HH);
        for (int i = tid; i < MT * HH / 4; i += 256) {
            int row = i / (HH / 4);
            int c4  = i - row * (HH / 4);
            dst[i] = ((const float4*)(emb + (size_t)fidx[row] * HH))[c4];
        }
    }

    // ---- pointer-head logits: 16 tiles of 16 rows of W_all[256,896] ----
    const int plen = *plen_p;
#pragma unroll 1
    for (int t = wave; t < NPTR / 16; t += 8) {
        const int j = t * 16 + ln;                 // 0..255 = p*128 + l
        v8f acc[4];
#pragma unroll
        for (int m = 0; m < 4; ++m)
#pragma unroll
            for (int i = 0; i < 8; ++i) acc[m][i] = 0.f;

        gemm_k(w_bf + (size_t)j * HH + half * 16, abase, acc);

        const int   l   = j & 127;
        const float add = bias[j] + ((l < plen) ? 0.f : -__builtin_inff());
#pragma unroll
        for (int m = 0; m < 4; ++m)
#pragma unroll
            for (int i = 0; i < 8; ++i) {
                int row = m * 16 + half * 8 + i;
                outB[(size_t)(r0 + row) * NPTR + j] = acc[m][i] + add;
            }
    }
}

extern "C" void kernel_launch(void* const* d_in, const int* in_sizes, int n_in,
                              void* d_out, int out_size, void* d_ws, size_t ws_size,
                              hipStream_t stream) {
    const float* z    = (const float*)d_in[0];
    const float* emb  = (const float*)d_in[1];
    const float* W    = (const float*)d_in[2];
    const float* b    = (const float*)d_in[3];
    const int*   plen = (const int*)d_in[4];

    // workspace layout: emb_bf (3,584,000 B) | w_bf (458,752 B) | hn (8,000 B)
    unsigned short* emb_bf = (unsigned short*)d_ws;
    unsigned short* w_bf   = (unsigned short*)((char*)d_ws + (size_t)KC * HH * 2);
    float*          hn     = (float*)((char*)d_ws + (size_t)KC * HH * 2 + (size_t)NPTR * HH * 2);

    convert_bf16<<<(KC * HH / 4 + 255) / 256, 256, 0, stream>>>(emb, emb_bf, KC * HH / 4);
    convert_bf16<<<(NPTR * HH / 4 + 255) / 256, 256, 0, stream>>>(W, w_bf, NPTR * HH / 4);
    half_norms<<<KC, 256, 0, stream>>>(emb, hn);

    float* outA = (float*)d_out;                 // op_state [B,H]
    float* outB = outA + (size_t)BB * HH;        // ptr_logits [B,2,128]
    vq_ptr_kernel<<<BB / MT, 256, 0, stream>>>(z, emb, emb_bf, hn, w_bf, b, plen,
                                               outA, outB);
}
